// SpatioTemporalGCN_91173565759517
// MI455X (gfx1250) — matrix operand order repaired and checked by reading, not compiled, not run on hardware
//
#include <hip/hip_runtime.h>
#include <hip/hip_bf16.h>
#include <math.h>

typedef __attribute__((ext_vector_type(16))) _Float16 v16h;
typedef __attribute__((ext_vector_type(8)))  _Float16 v8h;
typedef __attribute__((ext_vector_type(8)))  float    v8f;

#define N_NODES 100000
#define N_EDGES 1600000
#define HID 128
#define STATIC 16
#define DYN 8
#define TSTEPS 24
#define G4 512              // 4*HID
#define KCAT 160            // padded concat width for GCN layer 1 (16+128+16 pad)
#define KLSTM 160           // padded LSTM concat K: 128 h + 8 x + 24 pad
#define HSTRIDE 168         // LDS row stride for [h | x | pad]

// ---------------------------------------------------------------------------
// Packing / normalization prep kernels
// ---------------------------------------------------------------------------
// Wcat[512][160] = [ Whh (128) | Wih (8) | zeros (24) ]  in f16
__global__ void k_pack_lstm(const float* __restrict__ Whh, const float* __restrict__ Wih,
                            const float* __restrict__ bih, const float* __restrict__ bhh,
                            _Float16* __restrict__ wcat, float* __restrict__ bias,
                            float* __restrict__ deg) {
  int i = blockIdx.x * blockDim.x + threadIdx.x;
  if (i < G4 * KLSTM) {
    int r = i / KLSTM, k = i - r * KLSTM;
    float v;
    if (k < HID)            v = Whh[r * HID + k];
    else if (k < HID + DYN) v = Wih[r * DYN + (k - HID)];
    else                    v = 0.f;
    wcat[i] = (_Float16)v;
  }
  if (i < G4)      bias[i] = bih[i] + bhh[i];
  if (i < N_NODES) deg[i]  = 1.0f;     // self-loop weight
}

// transpose W[K][128] -> WT[128][Kpad] (f16, zero-padded K)
__global__ void k_pack_wt(const float* __restrict__ W, _Float16* __restrict__ WT,
                          int K, int Kpad) {
  int i = blockIdx.x * blockDim.x + threadIdx.x;
  if (i < HID * Kpad) {
    int j = i / Kpad, k = i - j * Kpad;
    WT[i] = (k < K) ? (_Float16)W[k * HID + j] : (_Float16)0.f;
  }
}

__global__ void k_deg(const float* __restrict__ ew, const int* __restrict__ ei,
                      float* __restrict__ deg) {
  int e = blockIdx.x * blockDim.x + threadIdx.x;
  if (e < N_EDGES) atomicAdd(&deg[ei[N_EDGES + e]], ew[e]);
}

__global__ void k_dis(const float* __restrict__ deg, float* __restrict__ dis) {
  int n = blockIdx.x * blockDim.x + threadIdx.x;
  if (n < N_NODES) {
    float d = deg[n];
    dis[n] = (d > 0.f) ? rsqrtf(fmaxf(d, 1e-12f)) : 0.f;
  }
}

__global__ void k_norm(const float* __restrict__ ew, const int* __restrict__ ei,
                       const float* __restrict__ dis, float* __restrict__ nrm) {
  int e = blockIdx.x * blockDim.x + threadIdx.x;
  if (e < N_EDGES) nrm[e] = dis[ei[e]] * ew[e] * dis[ei[N_EDGES + e]];
}

// ---------------------------------------------------------------------------
// LSTM: 32 nodes per block, 8 waves. Wave w owns hidden cols [16w,16w+16)
// for all 4 gates -> gate fusion is lane-local.
// z = [h | x_t] @ Wcat^T entirely via WMMA f16 (K = 160, 5 chunks).
// ---------------------------------------------------------------------------
__global__ __launch_bounds__(256) void k_lstm(
    const float* __restrict__ xs, const float* __restrict__ xd,
    const _Float16* __restrict__ wcat, const float* __restrict__ bias,
    _Float16* __restrict__ xcat) {
  __shared__ _Float16 sh_x[32 * TSTEPS * DYN];   // [m][t][d] f16, 12 KB
  __shared__ _Float16 sh_h[32 * HSTRIDE];        // [m][ h(128) | x(8) | pad ] 10.5 KB

  const int tid  = threadIdx.x;
  const int lane = tid & 31;
  const int wv   = tid >> 5;            // wave 0..7
  const int l16  = lane & 15;
  const int hi   = (lane >> 4) & 1;
  const int n0   = blockIdx.x * 32;

  // stage x_dynamic tile as f16; zero the [h|x|pad] tile
  for (int i = tid; i < 32 * TSTEPS * DYN; i += 256) {
    int m = i / (TSTEPS * DYN);
    int r = i - m * (TSTEPS * DYN);
    float v = (n0 + m < N_NODES) ? xd[(size_t)(n0 + m) * (TSTEPS * DYN) + r] : 0.f;
    sh_x[i] = (_Float16)v;
  }
  for (int i = tid; i < 32 * HSTRIDE; i += 256) sh_h[i] = (_Float16)0.f;

  const int hh = wv * 16 + l16;         // this lane's hidden unit
  float br[4];
#pragma unroll
  for (int G = 0; G < 4; ++G) br[G] = bias[G * HID + hh];

  float c[2][8];
#pragma unroll
  for (int mt = 0; mt < 2; ++mt)
#pragma unroll
    for (int v = 0; v < 8; ++v) c[mt][v] = 0.f;

  __syncthreads();

  // place x_0 into the x-slot of the concat tile
  {
    int m = tid >> 3, d = tid & 7;
    sh_h[m * HSTRIDE + HID + d] = sh_x[(m * TSTEPS + 0) * DYN + d];
  }
  __syncthreads();

  for (int t = 0; t < TSTEPS; ++t) {
    // ---- accumulator init: combined bias broadcast ----
    v8f acc[2][4];
#pragma unroll
    for (int mt = 0; mt < 2; ++mt)
#pragma unroll
      for (int G = 0; G < 4; ++G)
#pragma unroll
        for (int v = 0; v < 8; ++v) acc[mt][G][v] = br[G];

    // ---- [h | x_t] @ Wcat^T via WMMA: 2 M-tiles x 4 gates x 5 K-chunks ----
#pragma unroll
    for (int kc = 0; kc < 5; ++kc) {
      const int kbA = kc * 32 + hi * 16;       // A-layout K base
      v16h a[2];
#pragma unroll
      for (int mt = 0; mt < 2; ++mt) {
        const _Float16* hp = &sh_h[(mt * 16 + l16) * HSTRIDE];
        v8h lo = *(const v8h*)(hp + kbA);
        v8h ha = *(const v8h*)(hp + kbA + 8);
#pragma unroll
        for (int j = 0; j < 8; ++j) { a[mt][j] = lo[j]; a[mt][j + 8] = ha[j]; }
      }
      const int kbB = kc * 32 + hi * 16;       // B-layout K base
#pragma unroll
      for (int G = 0; G < 4; ++G) {
        int r = G * HID + hh;
        v16h b = *(const v16h*)(wcat + (size_t)r * KLSTM + kbB);
#pragma unroll
        for (int mt = 0; mt < 2; ++mt)
          acc[mt][G] = __builtin_amdgcn_wmma_f32_16x16x32_f16(
              false, a[mt], false, b, (short)0, acc[mt][G], false, false);
      }
    }

    __syncthreads();   // everyone done reading old [h | x_t]

    // ---- lane-local gate fusion + state update ----
#pragma unroll
    for (int mt = 0; mt < 2; ++mt) {
#pragma unroll
      for (int v = 0; v < 8; ++v) {
        float ig = 1.f / (1.f + __expf(-acc[mt][0][v]));
        float fg = 1.f / (1.f + __expf(-acc[mt][1][v]));
        float gg = tanhf(acc[mt][2][v]);
        float og = 1.f / (1.f + __expf(-acc[mt][3][v]));
        float cn = fg * c[mt][v] + ig * gg;
        c[mt][v] = cn;
        float hn = og * tanhf(cn);
        int m = mt * 16 + v + hi * 8;
        sh_h[m * HSTRIDE + hh] = (_Float16)hn;
      }
    }
    // refresh x-slot with x_{t+1}
    if (t + 1 < TSTEPS) {
      int m = tid >> 3, d = tid & 7;
      sh_h[m * HSTRIDE + HID + d] = sh_x[(m * TSTEPS + (t + 1)) * DYN + d];
    }
    __syncthreads();
  }

  // ---- write concat [static(16) | h(128) | pad(16)] as f16, stride 160 ----
  for (int i = tid; i < 32 * KCAT; i += 256) {
    int m = i / KCAT, col = i - m * KCAT;
    int n = n0 + m;
    if (n < N_NODES) {
      _Float16 v;
      if (col < STATIC)            v = (_Float16)xs[(size_t)n * STATIC + col];
      else if (col < STATIC + HID) v = sh_h[m * HSTRIDE + (col - STATIC)];
      else                         v = (_Float16)0.f;
      xcat[(size_t)n * KCAT + col] = v;
    }
  }
}

// ---------------------------------------------------------------------------
// GCN dense GEMM: Y[N,128] = X[N,Kpad](f16) @ WT[128,Kpad]^T via WMMA
// ---------------------------------------------------------------------------
__global__ __launch_bounds__(256) void k_gemm(
    const _Float16* __restrict__ X, int xstride, int nkc,
    const _Float16* __restrict__ WT, float* __restrict__ Y) {
  const int tid  = threadIdx.x;
  const int lane = tid & 31;
  const int wv   = tid >> 5;
  const int l16  = lane & 15;
  const int hi   = (lane >> 4) & 1;
  const int n0   = blockIdx.x * 32;

  v8f acc[2];
#pragma unroll
  for (int mt = 0; mt < 2; ++mt)
#pragma unroll
    for (int v = 0; v < 8; ++v) acc[mt][v] = 0.f;

  const int jcol = wv * 16 + l16;

  for (int kc = 0; kc < nkc; ++kc) {
    const int kbA = kc * 32 + hi * 16;
    v16h a[2];
#pragma unroll
    for (int mt = 0; mt < 2; ++mt) {
      int n = n0 + mt * 16 + l16;
      if (n >= N_NODES) n = N_NODES - 1;       // clamp (writes are guarded)
      const _Float16* xp = X + (size_t)n * xstride;
      v8h lo = *(const v8h*)(xp + kbA);
      v8h ha = *(const v8h*)(xp + kbA + 8);
#pragma unroll
      for (int j = 0; j < 8; ++j) { a[mt][j] = lo[j]; a[mt][j + 8] = ha[j]; }
    }
    v16h b = *(const v16h*)(WT + (size_t)jcol * xstride + kc * 32 + hi * 16);
#pragma unroll
    for (int mt = 0; mt < 2; ++mt)
      acc[mt] = __builtin_amdgcn_wmma_f32_16x16x32_f16(
          false, a[mt], false, b, (short)0, acc[mt], false, false);
  }

#pragma unroll
  for (int mt = 0; mt < 2; ++mt)
#pragma unroll
    for (int v = 0; v < 8; ++v) {
      int n = n0 + mt * 16 + v + hi * 8;
      if (n < N_NODES) Y[(size_t)n * HID + jcol] = acc[mt][v];
    }
}

// out[n][j] = bias[j] + dis[n]^2 * Y[n][j]   (self-loop term + bias init)
__global__ void k_selfbias(const float* __restrict__ Y, const float* __restrict__ dis,
                           const float* __restrict__ b, float* __restrict__ accb) {
  size_t i = (size_t)blockIdx.x * blockDim.x + threadIdx.x;
  if (i < (size_t)N_NODES * HID) {
    int n = (int)(i >> 7), j = (int)(i & 127);
    float d = dis[n];
    accb[i] = b[j] + d * d * Y[i];
  }
}

// scatter-add over edges: 32 threads per edge, 4 floats each
__global__ void k_scatter(const float* __restrict__ Y, const float* __restrict__ nrm,
                          const int* __restrict__ ei, float* __restrict__ accb) {
  long long tid = (long long)blockIdx.x * blockDim.x + threadIdx.x;
  long long e   = tid >> 5;
  int q         = (int)(tid & 31);
  if (e < N_EDGES) {
    int r = ei[e], cI = ei[N_EDGES + e];
    float nm = nrm[e];
    float4 yv = *(const float4*)(Y + (size_t)r * HID + q * 4);
    float* op = accb + (size_t)cI * HID + q * 4;
    atomicAdd(op + 0, nm * yv.x);
    atomicAdd(op + 1, nm * yv.y);
    atomicAdd(op + 2, nm * yv.z);
    atomicAdd(op + 3, nm * yv.w);
  }
}

__global__ void k_relupack(const float* __restrict__ accb, _Float16* __restrict__ X2) {
  size_t i = (size_t)blockIdx.x * blockDim.x + threadIdx.x;
  if (i < (size_t)N_NODES * HID) X2[i] = (_Float16)fmaxf(accb[i], 0.f);
}

// output head: one wave per node, relu(acc) . W_out + b_out
__global__ __launch_bounds__(256) void k_head(const float* __restrict__ accb,
                                              const float* __restrict__ wout,
                                              const float* __restrict__ bout,
                                              float* __restrict__ out) {
  int lane = threadIdx.x & 31;
  int wv   = threadIdx.x >> 5;
  int n    = blockIdx.x * 8 + wv;
  if (n >= N_NODES) return;
  float s = 0.f;
  for (int j = lane; j < HID; j += 32)
    s += fmaxf(accb[(size_t)n * HID + j], 0.f) * wout[j];
#pragma unroll
  for (int off = 16; off; off >>= 1) s += __shfl_xor(s, off, 32);
  if (lane == 0) out[n] = s + bout[0];
}

// ---------------------------------------------------------------------------
extern "C" void kernel_launch(void* const* d_in, const int* in_sizes, int n_in,
                              void* d_out, int out_size, void* d_ws, size_t ws_size,
                              hipStream_t stream) {
  (void)in_sizes; (void)n_in; (void)out_size; (void)ws_size;
  const float* xs   = (const float*)d_in[0];
  const float* xd   = (const float*)d_in[1];
  const int*   ei   = (const int*)d_in[2];
  const float* ew   = (const float*)d_in[3];
  const float* Wih  = (const float*)d_in[4];
  const float* Whh  = (const float*)d_in[5];
  const float* bih  = (const float*)d_in[6];
  const float* bhh  = (const float*)d_in[7];
  const float* W1   = (const float*)d_in[8];
  const float* b1   = (const float*)d_in[9];
  const float* W2   = (const float*)d_in[10];
  const float* b2   = (const float*)d_in[11];
  const float* W3   = (const float*)d_in[12];
  const float* b3   = (const float*)d_in[13];
  const float* Wout = (const float*)d_in[14];
  const float* bout = (const float*)d_in[15];
  float* out = (float*)d_out;

  // workspace carve-up (256B aligned)
  char* base = (char*)d_ws;
  size_t off = 0;
  auto alloc = [&](size_t bytes) -> char* {
    char* p = base + off;
    off += (bytes + 255) & ~(size_t)255;
    return p;
  };
  _Float16* wcat  = (_Float16*)alloc((size_t)G4 * KLSTM * 2);
  float*    bias  = (float*)alloc((size_t)G4 * 4);
  _Float16* w1t   = (_Float16*)alloc((size_t)HID * KCAT * 2);
  _Float16* w2t   = (_Float16*)alloc((size_t)HID * HID * 2);
  _Float16* w3t   = (_Float16*)alloc((size_t)HID * HID * 2);
  float*    deg   = (float*)alloc((size_t)N_NODES * 4);
  float*    dis   = (float*)alloc((size_t)N_NODES * 4);
  float*    nrm   = (float*)alloc((size_t)N_EDGES * 4);
  _Float16* xcat  = (_Float16*)alloc((size_t)N_NODES * KCAT * 2);
  _Float16* x2    = (_Float16*)alloc((size_t)N_NODES * HID * 2);
  float*    ybuf  = (float*)alloc((size_t)N_NODES * HID * 4);
  float*    accb  = (float*)alloc((size_t)N_NODES * HID * 4);

  const int B = 256;
  // --- prep ---
  k_pack_lstm<<<(N_NODES + B - 1) / B, B, 0, stream>>>(Whh, Wih, bih, bhh,
                                                       wcat, bias, deg);
  k_pack_wt<<<(HID * KCAT + B - 1) / B, B, 0, stream>>>(W1, w1t, STATIC + HID, KCAT);
  k_pack_wt<<<(HID * HID + B - 1) / B, B, 0, stream>>>(W2, w2t, HID, HID);
  k_pack_wt<<<(HID * HID + B - 1) / B, B, 0, stream>>>(W3, w3t, HID, HID);
  k_deg<<<(N_EDGES + B - 1) / B, B, 0, stream>>>(ew, ei, deg);
  k_dis<<<(N_NODES + B - 1) / B, B, 0, stream>>>(deg, dis);
  k_norm<<<(N_EDGES + B - 1) / B, B, 0, stream>>>(ew, ei, dis, nrm);

  // --- LSTM temporal encoder (WMMA) ---
  const int nb32 = (N_NODES + 31) / 32;
  k_lstm<<<nb32, B, 0, stream>>>(xs, xd, wcat, bias, xcat);

  const size_t nh = (size_t)N_NODES * HID;
  const int gElem = (int)((nh + B - 1) / B);
  const int gScat = (int)(((long long)N_EDGES * 32 + B - 1) / B);

  // --- GCN layer 1 ---
  k_gemm<<<nb32, B, 0, stream>>>(xcat, KCAT, 5, w1t, ybuf);
  k_selfbias<<<gElem, B, 0, stream>>>(ybuf, dis, b1, accb);
  k_scatter<<<gScat, B, 0, stream>>>(ybuf, nrm, ei, accb);
  k_relupack<<<gElem, B, 0, stream>>>(accb, x2);

  // --- GCN layer 2 ---
  k_gemm<<<nb32, B, 0, stream>>>(x2, HID, 4, w2t, ybuf);
  k_selfbias<<<gElem, B, 0, stream>>>(ybuf, dis, b2, accb);
  k_scatter<<<gScat, B, 0, stream>>>(ybuf, nrm, ei, accb);
  k_relupack<<<gElem, B, 0, stream>>>(accb, x2);

  // --- GCN layer 3 ---
  k_gemm<<<nb32, B, 0, stream>>>(x2, HID, 4, w3t, ybuf);
  k_selfbias<<<gElem, B, 0, stream>>>(ybuf, dis, b3, accb);
  k_scatter<<<gScat, B, 0, stream>>>(ybuf, nrm, ei, accb);

  // --- output head ---
  k_head<<<(N_NODES + 7) / 8, B, 0, stream>>>(accb, Wout, bout, out);
}
